// FlexSummaryAttention_14061722927969
// MI455X (gfx1250) — matrix operand order repaired and checked by reading, not compiled, not run on hardware
//
#include <hip/hip_runtime.h>
#include <stdint.h>

// ---------------- problem dims ----------------
#define BB    2
#define SS    64
#define RR    2048
#define DD    1024
#define HH    16
#define KVH_  4
#define HD_   64
#define GG    (HH / KVH_)      // 4
#define KVD_  (KVH_ * HD_)     // 256
#define LKTOT (SS + RR)        // 2112 keys, both phases
#define NCHUNK (LKTOT / 32)    // 66
#define SCALE_ 0.125f          // 64^-0.5
#define NEGINF_ (-1e9f)

typedef _Float16 v16h __attribute__((ext_vector_type(16)));
typedef _Float16 v8h  __attribute__((ext_vector_type(8)));
typedef float    v8f  __attribute__((ext_vector_type(8)));

__device__ __forceinline__ unsigned short f32_to_f16_raw(float f) {
  _Float16 h = (_Float16)f;
  return __builtin_bit_cast(unsigned short, h);
}

// ---- CDNA5 async global->LDS copy (ASYNCcnt-tracked DMA), 16B per lane ----
__device__ __forceinline__ void async_ld_b128(unsigned lds_off, const void* gptr) {
  asm volatile("global_load_async_to_lds_b128 %0, %1, off"
               :: "v"(lds_off), "v"((unsigned long long)(uintptr_t)gptr)
               : "memory");
}
__device__ __forceinline__ void wait_async0() {
  asm volatile("s_wait_asynccnt 0x0" ::: "memory");
}

// ---- fragment loaders (ISA 7.12.2 layouts, wave32) ----
// A-matrix 16x32 f16: lane row m = lane&15; halves e<8 -> k = kb+e,
// e>=8 -> k = 16+kb+(e-8), kb = (lane>>4)*8.  Two 16B chunks per lane.
__device__ __forceinline__ v16h load_afrag(const unsigned short* rowptr, int lane) {
  int kb = (lane >> 4) << 3;
  v8h lo = *(const v8h*)((const _Float16*)(rowptr + kb));
  v8h hi = *(const v8h*)((const _Float16*)(rowptr + 16 + kb));
  v16h f;
#pragma unroll
  for (int i = 0; i < 8; ++i) { f[i] = lo[i]; f[i + 8] = hi[i]; }
  return f;
}
// B-matrix 32x16 f16: lane col n = lane&15; same k order as A:
// e<8 -> k = kb+e, e>=8 -> k = 16+kb+(e-8), kb = (lane>>4)*8.  Source must be
// [n][k] (k contiguous) -> two 16B chunks per lane.
__device__ __forceinline__ v16h load_bfrag(const unsigned short* rowptr, int lane) {
  int kb = (lane >> 4) << 3;
  v8h lo = *(const v8h*)((const _Float16*)(rowptr + kb));
  v8h hi = *(const v8h*)((const _Float16*)(rowptr + 16 + kb));
  v16h f;
#pragma unroll
  for (int i = 0; i < 8; ++i) { f[i] = lo[i]; f[i + 8] = hi[i]; }
  return f;
}

// ---------------- GEMM: C[M,N] = A[M,K] * W^T[N,K]^T + bias ----------------
// Inputs pre-converted to f16; W pre-transposed to [N][K].  256 threads =
// 8 waves (4 M-tiles x 2 N-halves); block tile 64x128; wave tile 16x64
// (4 WMMA accumulators).  Double-buffered LDS filled by async DMA.
#define LPITCH 40   // halves (80B rows) -> 16B aligned, bank-staggered
__global__ __launch_bounds__(256)
void gemm_bias_kernel(const unsigned short* __restrict__ Ah,   // M x K f16
                      const unsigned short* __restrict__ WTh,  // N x K f16
                      const float* __restrict__ bias, float* __restrict__ C,
                      int M, int N, int K) {
  __shared__ unsigned short As[2][64 * LPITCH];
  __shared__ unsigned short Ws[2][128 * LPITCH];
  const int tid = threadIdx.x, lane = tid & 31, w = tid >> 5;
  const int wr = w & 3, wn = w >> 2;
  const int m0 = blockIdx.y * 64, n0 = blockIdx.x * 128;
  const int ln = lane & 15;
  (void)M;

  v8f acc[4];
  v8f z = {};
#pragma unroll
  for (int j = 0; j < 4; ++j) acc[j] = z;

  auto stage = [&](int k0, int p) {
    { // A tile 64x32: one b128 per thread
      int row = tid >> 2, ch = tid & 3;
      const unsigned short* g = Ah + (size_t)(m0 + row) * K + k0 + ch * 8;
      async_ld_b128((unsigned)(uintptr_t)&As[p][row * LPITCH + ch * 8], g);
    }
    { // W tile 128x32 (already [n][k]): two b128 per thread
      int rr = tid >> 1;
#pragma unroll
      for (int j = 0; j < 2; ++j) {
        int ch = (tid & 1) * 2 + j;
        const unsigned short* g = WTh + (size_t)(n0 + rr) * K + k0 + ch * 8;
        async_ld_b128((unsigned)(uintptr_t)&Ws[p][rr * LPITCH + ch * 8], g);
      }
    }
  };

  const int steps = K >> 5;
  stage(0, 0);
  for (int k = 0; k < steps; ++k) {
    const int p = k & 1;
    wait_async0();        // our fills of buffer p complete
    __syncthreads();      // everyone's fills complete; prev readers of 1-p done
    if (k + 1 < steps) stage((k + 1) << 5, p ^ 1);   // overlap next copy

    v16h af = load_afrag(&As[p][(wr * 16 + ln) * LPITCH], lane);
#pragma unroll
    for (int j = 0; j < 4; ++j) {
      v16h bf = load_bfrag(&Ws[p][(wn * 64 + j * 16 + ln) * LPITCH], lane);
      acc[j] = __builtin_amdgcn_wmma_f32_16x16x32_f16(false, af, false, bf, (short)0, acc[j], false, false);
    }
  }

#pragma unroll
  for (int j = 0; j < 4; ++j) {
    const int n_j = n0 + wn * 64 + j * 16 + ln;
    const float bj = bias[n_j];
#pragma unroll
    for (int r = 0; r < 8; ++r) {
      int m = m0 + wr * 16 + r + ((lane >> 4) << 3);
      C[(size_t)m * N + n_j] = acc[j][r] + bj;
    }
  }
}

// ---------------- elementwise conversions ----------------
__global__ void cvt_f16_kernel(const float* __restrict__ x, unsigned short* __restrict__ out, int total) {
  int i = blockIdx.x * blockDim.x + threadIdx.x;
  if (i < total) out[i] = f32_to_f16_raw(x[i]);
}

// W (K,N) f32 -> WT (N,K) f16 ; writes coalesced along k
__global__ void cvt_transpose_w_kernel(const float* __restrict__ w, unsigned short* __restrict__ wt,
                                       int K, int N) {
  int idx = blockIdx.x * blockDim.x + threadIdx.x;
  if (idx >= K * N) return;
  int k = idx % K, n = idx / K;
  wt[idx] = f32_to_f16_raw(w[(size_t)k * N + n]);
}

// V (B,L,KVD) f32 -> VT (B,KVH,HD,L) f16 ; writes coalesced along l
__global__ void cvt_transpose_v_kernel(const float* __restrict__ v, unsigned short* __restrict__ vt, int L) {
  int idx = blockIdx.x * blockDim.x + threadIdx.x;
  if (idx >= BB * KVH_ * HD_ * L) return;
  int l = idx % L;
  int d = (idx / L) % HD_;
  int kh = (idx / (L * HD_)) % KVH_;
  int b = idx / (L * HD_ * KVH_);
  vt[idx] = f32_to_f16_raw(v[(size_t)(b * L + l) * KVD_ + kh * HD_ + d]);
}

// ---------------- RoPE (1D summary / 2D hierarchical) + f16 cvt ----------------
__global__ void rope1d_kernel(const float* __restrict__ x, unsigned short* __restrict__ out,
                              int L, int nheads, int total) {
  int idx = blockIdx.x * blockDim.x + threadIdx.x;
  if (idx >= total) return;
  int d = idx & (HD_ - 1);
  int row = idx >> 6;
  int pos = (row / nheads) % L;
  int dd = d & 31;
  float ang = (float)pos * __expf(-(float)dd * 0.28782313662425572f);   // 10000^(-dd/32)
  float c = __cosf(ang), s = __sinf(ang);
  float xv = x[idx];
  float xp = x[(size_t)row * HD_ + ((d < 32) ? d + 32 : d - 32)];
  float r = (d < 32) ? (xv * c - xp * s) : (xv * c + xp * s);
  out[idx] = f32_to_f16_raw(r);
}

__global__ void rope2d_kernel(const float* __restrict__ x, unsigned short* __restrict__ out,
                              int L, int nheads, int total) {
  int idx = blockIdx.x * blockDim.x + threadIdx.x;
  if (idx >= total) return;
  int d = idx & (HD_ - 1);
  int row = idx >> 6;
  int pos = (row / nheads) % L;
  int bar = pos >> 5, tib = pos & 31;          // TPB = 32, analytic
  int dd = d & 31, j = dd & 15;
  float p = (dd < 16) ? (float)bar : (float)tib;
  float ang = p * __expf(-(float)j * 0.57564627324851144f);             // 10000^(-j/16)
  float c = __cosf(ang), s = __sinf(ang);
  float xv = x[idx];
  float xp = x[(size_t)row * HD_ + ((d < 32) ? d + 32 : d - 32)];
  float r = (d < 32) ? (xv * c - xp * s) : (xv * c + xp * s);
  out[idx] = f32_to_f16_raw(r);
}

// ---------------- analytic block masks ----------------
__device__ __forceinline__ bool attn_allowed(int q, int j, int phase) {
  if (phase == 0) {                        // summarize: key_bar[j] <= i
    int jb = (j < SS) ? j : ((j - SS) >> 5);
    return jb <= q;
  }                                        // updating: j<=bar(t) | s<=t
  return (j < SS) ? (j <= (q >> 5)) : ((j - SS) <= q);
}

// ---------------- fused flash attention (barrier-free) ----------------
// Block = 8 waves = 4 GQA q-heads x 2 query-tiles on one kv-head stream.
// K read row-major from global (B-frag = two 16B chunks/lane);
// V read from pre-transposed [b][kvh][d][L] global buffer the same way.
// Only wave-private LDS: P tile C-layout -> A-layout relayout.
#define VPITCH 40
__global__ __launch_bounds__(256)
void attn_kernel(const unsigned short* __restrict__ qbf,
                 const unsigned short* __restrict__ ksum, const unsigned short* __restrict__ kreg,
                 const unsigned short* __restrict__ vtsum, const unsigned short* __restrict__ vtreg,
                 float* __restrict__ out, int Lq, int phase) {
  __shared__ unsigned short Pb[8 * 16 * VPITCH];

  const int tid = threadIdx.x, lane = tid & 31, w = tid >> 5;
  const int nqb = Lq >> 5;
  int bid = blockIdx.x;
  const int qb = bid % nqb; bid /= nqb;
  const int kh = bid % KVH_; const int b = bid / KVH_;
  const int qt = w & 1, hq = w >> 1;
  const int qh = kh * GG + hq;
  const int q0 = qb * 32 + qt * 16;
  const int ln = lane & 15, hig = lane >> 4;

  const unsigned short* qrow = qbf + ((size_t)(b * Lq + q0 + ln) * DD + qh * HD_);
  v16h fq0 = load_afrag(qrow, lane);
  v16h fq1 = load_afrag(qrow + 32, lane);

  v8f o0 = {}, o1 = {}, o2 = {}, o3 = {};
  float mrow[8], lrow[8];
#pragma unroll
  for (int r = 0; r < 8; ++r) { mrow[r] = -3.0e38f; lrow[r] = 0.f; }

  unsigned short* Pw = &Pb[w * 16 * VPITCH];

  for (int kc = 0; kc < NCHUNK; ++kc) {
    const int keybase = kc * 32;
    // chunk is entirely inside one key segment (S = 64 = 2 chunks)
    const bool insum = keybase < SS;
    const unsigned short* kseg = insum ? ksum : kreg;
    const unsigned short* vseg = insum ? vtsum : vtreg;
    const int segL = insum ? SS : RR;
    const int krow0 = insum ? keybase : keybase - SS;

    if (kc + 1 < NCHUNK) {  // gfx1250 global_prefetch_b8 of next K chunk
      const bool ns = (keybase + 32) < SS;
      const unsigned short* pk = (ns ? ksum : kreg)
          + ((size_t)(b * (ns ? SS : RR) + (ns ? keybase + 32 : keybase + 32 - SS) + ln) * KVD_ + kh * HD_);
      __builtin_prefetch(pk, 0, 1);
    }

    // --- scores: S = Q * K^T
    v8f s0 = {}, s1 = {};
    {
      const unsigned short* krow = kseg + ((size_t)(b * segL + krow0 + ln) * KVD_ + kh * HD_);
      v16h fk0 = load_bfrag(krow, lane);
      v16h fk1 = load_bfrag(krow + 32, lane);
      s0 = __builtin_amdgcn_wmma_f32_16x16x32_f16(false, fq0, false, fk0, (short)0, s0, false, false);
      s0 = __builtin_amdgcn_wmma_f32_16x16x32_f16(false, fq1, false, fk1, (short)0, s0, false, false);
    }
    {
      const unsigned short* krow = kseg + ((size_t)(b * segL + krow0 + 16 + ln) * KVD_ + kh * HD_);
      v16h fk0 = load_bfrag(krow, lane);
      v16h fk1 = load_bfrag(krow + 32, lane);
      s1 = __builtin_amdgcn_wmma_f32_16x16x32_f16(false, fq0, false, fk0, (short)0, s1, false, false);
      s1 = __builtin_amdgcn_wmma_f32_16x16x32_f16(false, fq1, false, fk1, (short)0, s1, false, false);
    }

    // --- online softmax (row reductions across 16-lane groups)
    const int j0 = keybase + ln, j1 = j0 + 16;
#pragma unroll
    for (int r = 0; r < 8; ++r) {
      const int qi = q0 + r + (hig << 3);
      float a = s0[r] * SCALE_, c2 = s1[r] * SCALE_;
      if (!attn_allowed(qi, j0, phase)) a = NEGINF_;
      if (!attn_allowed(qi, j1, phase)) c2 = NEGINF_;
      float t = fmaxf(a, c2);
#pragma unroll
      for (int off = 1; off < 16; off <<= 1) t = fmaxf(t, __shfl_xor(t, off, 32));
      float mn = fmaxf(mrow[r], t);
      float p0 = __expf(a - mn), p1 = __expf(c2 - mn);
      float rs = p0 + p1;
#pragma unroll
      for (int off = 1; off < 16; off <<= 1) rs += __shfl_xor(rs, off, 32);
      float alpha = __expf(mrow[r] - mn);
      lrow[r] = lrow[r] * alpha + rs;
      mrow[r] = mn;
      o0[r] *= alpha; o1[r] *= alpha; o2[r] *= alpha; o3[r] *= alpha;
      int m = r + (hig << 3);
      Pw[m * VPITCH + ln]      = f32_to_f16_raw(p0);
      Pw[m * VPITCH + 16 + ln] = f32_to_f16_raw(p1);
    }

    // --- O += P * V  (P via wave-private LDS relayout; V pre-transposed)
    v16h fp = load_afrag(&Pw[ln * VPITCH], lane);
    const unsigned short* vb = vseg + ((size_t)((b * KVH_ + kh) * HD_) + ln) * segL + krow0;
    v16h fv;
    fv = load_bfrag(vb + (size_t)0  * segL, lane);
    o0 = __builtin_amdgcn_wmma_f32_16x16x32_f16(false, fp, false, fv, (short)0, o0, false, false);
    fv = load_bfrag(vb + (size_t)16 * segL, lane);
    o1 = __builtin_amdgcn_wmma_f32_16x16x32_f16(false, fp, false, fv, (short)0, o1, false, false);
    fv = load_bfrag(vb + (size_t)32 * segL, lane);
    o2 = __builtin_amdgcn_wmma_f32_16x16x32_f16(false, fp, false, fv, (short)0, o2, false, false);
    fv = load_bfrag(vb + (size_t)48 * segL, lane);
    o3 = __builtin_amdgcn_wmma_f32_16x16x32_f16(false, fp, false, fv, (short)0, o3, false, false);
  }

#pragma unroll
  for (int r = 0; r < 8; ++r) {
    float inv = 1.f / lrow[r];
    int m = q0 + r + (hig << 3);
    float* orow = out + ((size_t)(b * Lq + m) * DD + qh * HD_);
    orow[0  + ln] = o0[r] * inv;
    orow[16 + ln] = o1[r] * inv;
    orow[32 + ln] = o2[r] * inv;
    orow[48 + ln] = o3[r] * inv;
  }
}

// ---------------- host orchestration ----------------
extern "C" void kernel_launch(void* const* d_in, const int* in_sizes, int n_in,
                              void* d_out, int out_size, void* d_ws, size_t ws_size,
                              hipStream_t stream) {
  (void)in_sizes; (void)n_in; (void)out_size; (void)ws_size;
  const float* sum_x = (const float*)d_in[0];
  const float* reg_x = (const float*)d_in[1];
  const float* w_sq = (const float*)d_in[2];  const float* b_sq = (const float*)d_in[3];
  const float* w_sk = (const float*)d_in[4];  const float* b_sk = (const float*)d_in[5];
  const float* w_sv = (const float*)d_in[6];  const float* b_sv = (const float*)d_in[7];
  const float* w_so = (const float*)d_in[8];  const float* b_so = (const float*)d_in[9];
  const float* w_rq = (const float*)d_in[10]; const float* b_rq = (const float*)d_in[11];
  const float* w_rk = (const float*)d_in[12]; const float* b_rk = (const float*)d_in[13];
  const float* w_rv = (const float*)d_in[14]; const float* b_rv = (const float*)d_in[15];
  const float* w_ro = (const float*)d_in[16]; const float* b_ro = (const float*)d_in[17];
  const float* w_k2 = (const float*)d_in[18]; const float* b_k2 = (const float*)d_in[19];
  const float* w_v2 = (const float*)d_in[20]; const float* b_v2 = (const float*)d_in[21];
  // d_in[22..25] (bar ids, token-in-bar ids, masks) recomputed analytically on device.

  const int MS = BB * SS;   // 128
  const int MR = BB * RR;   // 4096

  char* ws = (char*)d_ws;
  size_t off = 0;
  auto alloc_f = [&](size_t n) { float* p = (float*)(ws + off); off += ((n * 4 + 255) / 256) * 256; return p; };
  auto alloc_h = [&](size_t n) { unsigned short* p = (unsigned short*)(ws + off); off += ((n * 2 + 255) / 256) * 256; return p; };

  // f32 intermediates
  float* sum_q_f = alloc_f((size_t)MS * DD);
  float* sum_k_f = alloc_f((size_t)MS * KVD_);
  float* sum_v_f = alloc_f((size_t)MS * KVD_);
  float* reg_q_f = alloc_f((size_t)MR * DD);
  float* reg_k_f = alloc_f((size_t)MR * KVD_);
  float* reg_v_f = alloc_f((size_t)MR * KVD_);
  float* sum_attn = alloc_f((size_t)MS * DD);
  float* reg_attn = alloc_f((size_t)MR * DD);
  float* k2_f = alloc_f((size_t)MS * KVD_);
  float* v2_f = alloc_f((size_t)MS * KVD_);
  // f16 activations / attention operands
  unsigned short* xs_h = alloc_h((size_t)MS * DD);
  unsigned short* xr_h = alloc_h((size_t)MR * DD);
  unsigned short* sum_q_h = alloc_h((size_t)MS * DD);
  unsigned short* sum_k_h = alloc_h((size_t)MS * KVD_);
  unsigned short* reg_q_h = alloc_h((size_t)MR * DD);
  unsigned short* reg_k_h = alloc_h((size_t)MR * KVD_);
  unsigned short* k2_h = alloc_h((size_t)MS * KVD_);
  unsigned short* vtsum_h = alloc_h((size_t)MS * KVD_);          // (B,KVH,HD,S)
  unsigned short* vtreg_h = alloc_h((size_t)MR * KVD_);          // (B,KVH,HD,R)
  unsigned short* vt2_h = alloc_h((size_t)MS * KVD_);            // (B,KVH,HD,S)
  unsigned short* sumattn_h = alloc_h((size_t)MS * DD);
  unsigned short* regattn_h = alloc_h((size_t)MR * DD);
  // f16 pre-transposed weights [N][K]
  unsigned short* wt_sq = alloc_h((size_t)DD * DD);
  unsigned short* wt_sk = alloc_h((size_t)DD * KVD_);
  unsigned short* wt_sv = alloc_h((size_t)DD * KVD_);
  unsigned short* wt_rq = alloc_h((size_t)DD * DD);
  unsigned short* wt_rk = alloc_h((size_t)DD * KVD_);
  unsigned short* wt_rv = alloc_h((size_t)DD * KVD_);
  unsigned short* wt_k2 = alloc_h((size_t)DD * KVD_);
  unsigned short* wt_v2 = alloc_h((size_t)DD * KVD_);
  unsigned short* wt_so = alloc_h((size_t)DD * DD);
  unsigned short* wt_ro = alloc_h((size_t)DD * DD);

  dim3 blk(256);
  auto g1 = [](int n) { return dim3((unsigned)((n + 255) / 256)); };

  // one-time conversions: activations to f16, weights to f16 [N][K]
  cvt_f16_kernel<<<g1(MS * DD), blk, 0, stream>>>(sum_x, xs_h, MS * DD);
  cvt_f16_kernel<<<g1(MR * DD), blk, 0, stream>>>(reg_x, xr_h, MR * DD);
  cvt_transpose_w_kernel<<<g1(DD * DD), blk, 0, stream>>>(w_sq, wt_sq, DD, DD);
  cvt_transpose_w_kernel<<<g1(DD * KVD_), blk, 0, stream>>>(w_sk, wt_sk, DD, KVD_);
  cvt_transpose_w_kernel<<<g1(DD * KVD_), blk, 0, stream>>>(w_sv, wt_sv, DD, KVD_);
  cvt_transpose_w_kernel<<<g1(DD * DD), blk, 0, stream>>>(w_rq, wt_rq, DD, DD);
  cvt_transpose_w_kernel<<<g1(DD * KVD_), blk, 0, stream>>>(w_rk, wt_rk, DD, KVD_);
  cvt_transpose_w_kernel<<<g1(DD * KVD_), blk, 0, stream>>>(w_rv, wt_rv, DD, KVD_);
  cvt_transpose_w_kernel<<<g1(DD * KVD_), blk, 0, stream>>>(w_k2, wt_k2, DD, KVD_);
  cvt_transpose_w_kernel<<<g1(DD * KVD_), blk, 0, stream>>>(w_v2, wt_v2, DD, KVD_);
  cvt_transpose_w_kernel<<<g1(DD * DD), blk, 0, stream>>>(w_so, wt_so, DD, DD);
  cvt_transpose_w_kernel<<<g1(DD * DD), blk, 0, stream>>>(w_ro, wt_ro, DD, DD);

  // projections (async-DMA double-buffered WMMA GEMMs)
  gemm_bias_kernel<<<dim3(DD / 128, MS / 64), blk, 0, stream>>>(xs_h, wt_sq, b_sq, sum_q_f, MS, DD, DD);
  gemm_bias_kernel<<<dim3(KVD_ / 128, MS / 64), blk, 0, stream>>>(xs_h, wt_sk, b_sk, sum_k_f, MS, KVD_, DD);
  gemm_bias_kernel<<<dim3(KVD_ / 128, MS / 64), blk, 0, stream>>>(xs_h, wt_sv, b_sv, sum_v_f, MS, KVD_, DD);
  gemm_bias_kernel<<<dim3(DD / 128, MR / 64), blk, 0, stream>>>(xr_h, wt_rq, b_rq, reg_q_f, MR, DD, DD);
  gemm_bias_kernel<<<dim3(KVD_ / 128, MR / 64), blk, 0, stream>>>(xr_h, wt_rk, b_rk, reg_k_f, MR, KVD_, DD);
  gemm_bias_kernel<<<dim3(KVD_ / 128, MR / 64), blk, 0, stream>>>(xr_h, wt_rv, b_rv, reg_v_f, MR, KVD_, DD);

  // RoPE + f16; V transposed to [b][kvh][d][L]
  rope1d_kernel<<<g1(MS * DD), blk, 0, stream>>>(sum_q_f, sum_q_h, SS, HH, MS * DD);
  rope1d_kernel<<<g1(MS * KVD_), blk, 0, stream>>>(sum_k_f, sum_k_h, SS, KVH_, MS * KVD_);
  rope2d_kernel<<<g1(MR * DD), blk, 0, stream>>>(reg_q_f, reg_q_h, RR, HH, MR * DD);
  rope2d_kernel<<<g1(MR * KVD_), blk, 0, stream>>>(reg_k_f, reg_k_h, RR, KVH_, MR * KVD_);
  cvt_transpose_v_kernel<<<g1(MS * KVD_), blk, 0, stream>>>(sum_v_f, vtsum_h, SS);
  cvt_transpose_v_kernel<<<g1(MR * KVD_), blk, 0, stream>>>(reg_v_f, vtreg_h, RR);

  // phase 1: summarize
  attn_kernel<<<BB * KVH_ * (SS / 32), blk, 0, stream>>>(sum_q_h, sum_k_h, reg_k_h, vtsum_h, vtreg_h, sum_attn, SS, 0);

  // K2/V2 projections of summaries
  cvt_f16_kernel<<<g1(MS * DD), blk, 0, stream>>>(sum_attn, sumattn_h, MS * DD);
  gemm_bias_kernel<<<dim3(KVD_ / 128, MS / 64), blk, 0, stream>>>(sumattn_h, wt_k2, b_k2, k2_f, MS, KVD_, DD);
  gemm_bias_kernel<<<dim3(KVD_ / 128, MS / 64), blk, 0, stream>>>(sumattn_h, wt_v2, b_v2, v2_f, MS, KVD_, DD);
  cvt_f16_kernel<<<g1(MS * KVD_), blk, 0, stream>>>(k2_f, k2_h, MS * KVD_);
  cvt_transpose_v_kernel<<<g1(MS * KVD_), blk, 0, stream>>>(v2_f, vt2_h, SS);

  // phase 2: updating
  attn_kernel<<<BB * KVH_ * (RR / 32), blk, 0, stream>>>(reg_q_h, k2_h, reg_k_h, vt2_h, vtreg_h, reg_attn, RR, 1);

  // output projections straight into d_out (tuple order: sum, reg)
  cvt_f16_kernel<<<g1(MR * DD), blk, 0, stream>>>(reg_attn, regattn_h, MR * DD);
  float* out_sum = (float*)d_out;
  float* out_reg = out_sum + (size_t)MS * DD;
  gemm_bias_kernel<<<dim3(DD / 128, MS / 64), blk, 0, stream>>>(sumattn_h, wt_so, b_so, out_sum, MS, DD, DD);
  gemm_bias_kernel<<<dim3(DD / 128, MR / 64), blk, 0, stream>>>(regattn_h, wt_ro, b_ro, out_reg, MR, DD, DD);
}
